// GraphTransformerNetwork_24859270709325
// MI455X (gfx1250) — matrix operand order repaired and checked by reading, not compiled, not run on hardware
//
#include <hip/hip_runtime.h>
#include <hip/hip_bf16.h>
#include <stdint.h>

#define N_NODES 20000
#define E_EDGES 640000
#define DIM 128
#define LAYERS 6
#define NCLASS 20
#define NEG_SLOPE 0.2f
#define EPS_F 1e-16f

typedef __bf16 bf16_t;
typedef __attribute__((ext_vector_type(16))) __bf16 v16bf;
typedef __attribute__((ext_vector_type(8)))  __bf16 v8bf;
typedef __attribute__((ext_vector_type(8)))  float  v8f;

// Load a 16-element bf16 fragment from two 16B-aligned 8-element chunks.
static __device__ __forceinline__ v16bf load_frag(const bf16_t* p0, const bf16_t* p1) {
  v8bf lo = *(const v8bf*)p0;
  v8bf hi = *(const v8bf*)p1;
  v16bf r;
#pragma unroll
  for (int i = 0; i < 8; ++i) { r[i] = lo[i]; r[i + 8] = hi[i]; }
  return r;
}

static __device__ __forceinline__ v8f wmma_bf16(v16bf a, v16bf b, v8f c) {
  return __builtin_amdgcn_wmma_f32_16x16x32_bf16(false, a, false, b, (short)0, c, false, false);
}

// ---------------- prep kernels ----------------

__global__ void k_f32_to_bf16(const float* __restrict__ in, bf16_t* __restrict__ out, long n) {
  long i = (long)blockIdx.x * blockDim.x + threadIdx.x;
  if (i < n) out[i] = (bf16_t)in[i];
}

__global__ void k_zero_u128(uint4* __restrict__ p, long n4) {
  long i = (long)blockIdx.x * blockDim.x + threadIdx.x;
  if (i < n4) p[i] = make_uint4(0u, 0u, 0u, 0u);
}

// Fold edge weights: Wc = Wee @ We (store transposed bf16), bc = bee @ We.
// Also transpose+convert Wl, Wr, Wlin to [n][k] bf16 layout.
__global__ void k_prep_weights(const float* __restrict__ Wee, const float* __restrict__ bee,
                               const float* __restrict__ Wl,  const float* __restrict__ Wr,
                               const float* __restrict__ Wlin, const float* __restrict__ We,
                               bf16_t* __restrict__ Wct, float* __restrict__ bc,
                               bf16_t* __restrict__ Wlt, bf16_t* __restrict__ Wrt,
                               bf16_t* __restrict__ Wlint) {
  int idx = blockIdx.x * blockDim.x + threadIdx.x;
  if (idx >= LAYERS * DIM * DIM) return;
  int l = idx / (DIM * DIM);
  int rem = idx - l * DIM * DIM;
  int n = rem / DIM, k = rem - n * DIM;
  const float* WeeL = Wee + (size_t)l * DIM * DIM;
  const float* WeL  = We  + (size_t)l * DIM * DIM;
  float s = 0.f;
  for (int j = 0; j < DIM; ++j) s += WeeL[k * DIM + j] * WeL[j * DIM + n];
  size_t o = (size_t)l * DIM * DIM + n * DIM + k;
  size_t src_o = (size_t)l * DIM * DIM + k * DIM + n;
  Wct[o]   = (bf16_t)s;
  Wlt[o]   = (bf16_t)Wl[src_o];
  Wrt[o]   = (bf16_t)Wr[src_o];
  Wlint[o] = (bf16_t)Wlin[src_o];
  if (k == 0) {
    float b = 0.f;
    for (int j = 0; j < DIM; ++j) b += bee[l * DIM + j] * WeL[j * DIM + n];
    bc[l * DIM + n] = b;
  }
}

// ---------------- WMMA GEMM: C[M,128] = A[M,128](bf16) @ B(Bt[n][k] bf16) + bias ----------------
// grid.x = M/16 row tiles, grid.y = 8 col tiles, block = 32 (one wave).
// DUAL: additionally emit a bf16 copy of C (next layer's activation input).

template <bool DUAL>
__global__ __launch_bounds__(32)
void k_gemm128(const bf16_t* __restrict__ A, const bf16_t* __restrict__ Bt,
               const float* __restrict__ bias,
               float* __restrict__ Cf, bf16_t* __restrict__ Cb) {
  int lane = threadIdx.x;
  int r = lane & 15, half = lane >> 4, kb = half * 8;
  int m0 = blockIdx.x * 16, n0 = blockIdx.y * 16;
  const bf16_t* arow = A + (size_t)(m0 + r) * DIM;
  const bf16_t* brow = Bt + (size_t)(n0 + r) * DIM;
  if ((int)blockIdx.x + 1 < (int)gridDim.x)
    __builtin_prefetch(arow + 16 * DIM, 0, 0);
  float bv = bias[n0 + r];
  v8f acc = {bv, bv, bv, bv, bv, bv, bv, bv};
#pragma unroll
  for (int ks = 0; ks < 4; ++ks) {
    v16bf a = load_frag(arow + ks * 32 + kb, arow + ks * 32 + 16 + kb);
    v16bf b = load_frag(brow + ks * 32 + kb, brow + ks * 32 + 16 + kb);
    acc = wmma_bf16(a, b, acc);
  }
#pragma unroll
  for (int v = 0; v < 8; ++v) {
    size_t o = (size_t)(m0 + v + 8 * half) * DIM + n0 + r;
    Cf[o] = acc[v];
    if (DUAL) Cb[o] = (bf16_t)acc[v];
  }
}

// ---------------- attention pass 1: fused ee-GEMM + logits + segment max ----------------
// One wave per 16-edge tile: 32 WMMAs compute the 16x128 edge-embedding tile,
// gathers of xl[src]/xr[dst] come from L2 (node tensors fit in 192MB L2),
// per-edge/head reduction via LDS float atomics, segment max via monotonic-uint
// global atomicMax.

__global__ __launch_bounds__(32)
void k_attn_logits(const bf16_t* __restrict__ ea, const bf16_t* __restrict__ Wct,
                   const float* __restrict__ bc,
                   const float* __restrict__ xl, const float* __restrict__ xr,
                   const int* __restrict__ src, const int* __restrict__ dst,
                   const float* __restrict__ att,
                   float* __restrict__ logits, unsigned* __restrict__ amax_key) {
  __shared__ int s_src[16], s_dst[16];
  __shared__ float s_log[64];  // 16 edges x 4 heads
  int lane = threadIdx.x;
  int e0 = blockIdx.x * 16;
  if (lane < 16) { s_src[lane] = src[e0 + lane]; s_dst[lane] = dst[e0 + lane]; }
  s_log[lane] = 0.f; s_log[lane + 32] = 0.f;
  __syncthreads();

  int r = lane & 15, half = lane >> 4, kb = half * 8;
  const bf16_t* arow = ea + (size_t)(e0 + r) * DIM;
  if ((int)blockIdx.x + 1 < (int)gridDim.x)
    __builtin_prefetch(arow + 16 * DIM, 0, 0);   // stream next edge tile

  v16bf afr[4];
#pragma unroll
  for (int ks = 0; ks < 4; ++ks)
    afr[ks] = load_frag(arow + ks * 32 + kb, arow + ks * 32 + 16 + kb);

#pragma unroll
  for (int h = 0; h < 4; ++h) {
    float part[8];
#pragma unroll
    for (int v = 0; v < 8; ++v) part[v] = 0.f;
#pragma unroll
    for (int sub = 0; sub < 2; ++sub) {
      int nt = h * 2 + sub;
      int ch = nt * 16 + r;
      float attv = att[ch];
      float bcv = bc[ch];
      v8f acc = {bcv, bcv, bcv, bcv, bcv, bcv, bcv, bcv};
      const bf16_t* brow = Wct + (size_t)ch * DIM;
#pragma unroll
      for (int ks = 0; ks < 4; ++ks) {
        v16bf b = load_frag(brow + ks * 32 + kb, brow + ks * 32 + 16 + kb);
        acc = wmma_bf16(afr[ks], b, acc);
      }
#pragma unroll
      for (int v = 0; v < 8; ++v) {
        int e = v + 8 * half;
        float mv = acc[v] + xl[(size_t)s_src[e] * DIM + ch] + xr[(size_t)s_dst[e] * DIM + ch];
        mv = mv > 0.f ? mv : mv * NEG_SLOPE;
        part[v] += attv * mv;
      }
    }
#pragma unroll
    for (int v = 0; v < 8; ++v)
      atomicAdd(&s_log[(v + 8 * half) * 4 + h], part[v]);
  }
  __syncthreads();

#pragma unroll
  for (int i = lane; i < 64; i += 32) {
    int e = i >> 2, h = i & 3;
    float lg = s_log[i];
    logits[(size_t)(e0 + e) * 4 + h] = lg;
    unsigned u = __float_as_uint(lg);
    unsigned key = (u & 0x80000000u) ? ~u : (u | 0x80000000u);  // order-preserving
    atomicMax(&amax_key[(size_t)s_dst[e] * 4 + h], key);
  }
}

// ---------------- attention pass 2: exp + segment sum (one thread per edge) ----------------

__global__ void k_attn_exp(const float* __restrict__ logits, const int* __restrict__ dst,
                           const unsigned* __restrict__ amax_key,
                           float* __restrict__ denom, float* __restrict__ exs) {
  int e = blockIdx.x * blockDim.x + threadIdx.x;
  if (e >= E_EDGES) return;
  int d = dst[e];
  float4 lg = *(const float4*)(logits + (size_t)e * 4);
  uint4  k4 = *(const uint4*)(amax_key + (size_t)d * 4);
  unsigned ks[4] = {k4.x, k4.y, k4.z, k4.w};
  float lgv[4] = {lg.x, lg.y, lg.z, lg.w};
  float ex[4];
#pragma unroll
  for (int h = 0; h < 4; ++h) {
    unsigned u = (ks[h] & 0x80000000u) ? (ks[h] & 0x7fffffffu) : ~ks[h];
    ex[h] = __expf(lgv[h] - __uint_as_float(u));
    atomicAdd(&denom[(size_t)d * 4 + h], ex[h]);
  }
  *(float4*)(exs + (size_t)e * 4) = make_float4(ex[0], ex[1], ex[2], ex[3]);
}

// ---------------- attention pass 3: weighted scatter (one thread per edge-head) ----------------

__global__ void k_attn_scatter(const float* __restrict__ exs, const float* __restrict__ denom,
                               const float* __restrict__ xl,
                               const int* __restrict__ src, const int* __restrict__ dst,
                               float* __restrict__ agg) {
  long idx = (long)blockIdx.x * blockDim.x + threadIdx.x;
  if (idx >= (long)E_EDGES * 4) return;
  int e = (int)(idx >> 2), h = (int)(idx & 3);
  int d = dst[e], s = src[e];
  float alpha = exs[(size_t)e * 4 + h] / (denom[(size_t)d * 4 + h] + EPS_F);
  const float4* xs = (const float4*)(xl + (size_t)s * DIM + h * 32);
  float* ag = agg + (size_t)d * DIM + h * 32;
#pragma unroll
  for (int q = 0; q < 8; ++q) {
    float4 v = xs[q];
    atomicAdd(ag + q * 4 + 0, v.x * alpha);
    atomicAdd(ag + q * 4 + 1, v.y * alpha);
    atomicAdd(ag + q * 4 + 2, v.z * alpha);
    atomicAdd(ag + q * 4 + 3, v.w * alpha);
  }
}

// ---------------- relu(agg + cb) -> bf16 activation ----------------

__global__ void k_act(const float* __restrict__ agg, const float* __restrict__ cb,
                      bf16_t* __restrict__ out) {
  long i = (long)blockIdx.x * blockDim.x + threadIdx.x;
  if (i >= (long)N_NODES * DIM) return;
  float v = agg[i] + cb[(int)(i & 127)];
  out[i] = (bf16_t)(v > 0.f ? v : 0.f);
}

// ---------------- final classifier: h @ Wout + bout ----------------

__global__ void k_classifier(const float* __restrict__ h, const float* __restrict__ Wout,
                             const float* __restrict__ bout, float* __restrict__ out) {
  long idx = (long)blockIdx.x * blockDim.x + threadIdx.x;
  if (idx >= (long)N_NODES * NCLASS) return;
  int n = (int)(idx / NCLASS), c = (int)(idx - (long)n * NCLASS);
  float s = bout[c];
  const float* hr = h + (size_t)n * DIM;
#pragma unroll 4
  for (int k = 0; k < DIM; ++k) s += hr[k] * Wout[k * NCLASS + c];
  out[idx] = s;
}

// ---------------- host launch ----------------

extern "C" void kernel_launch(void* const* d_in, const int* in_sizes, int n_in,
                              void* d_out, int out_size, void* d_ws, size_t ws_size,
                              hipStream_t stream) {
  (void)in_sizes; (void)n_in; (void)out_size; (void)ws_size;
  const float* x         = (const float*)d_in[0];
  const int*   edge_idx  = (const int*)d_in[1];
  const int*   src       = edge_idx;
  const int*   dst       = edge_idx + E_EDGES;
  const float* edge_attr = (const float*)d_in[2];
  const float* Wee  = (const float*)d_in[3];
  const float* bee  = (const float*)d_in[4];
  const float* Wl   = (const float*)d_in[5];
  const float* bl   = (const float*)d_in[6];
  const float* Wr   = (const float*)d_in[7];
  const float* br   = (const float*)d_in[8];
  const float* We   = (const float*)d_in[9];
  const float* att  = (const float*)d_in[10];
  const float* cb   = (const float*)d_in[11];
  const float* Wlin = (const float*)d_in[12];
  const float* blin = (const float*)d_in[13];
  const float* Wout = (const float*)d_in[14];
  const float* bout = (const float*)d_in[15];
  float* out = (float*)d_out;

  char* ws = (char*)d_ws;
  size_t off = 0;
  auto alloc = [&](size_t bytes) -> void* {
    void* p = ws + off;
    off = (off + bytes + 255) & ~(size_t)255;
    return p;
  };
  bf16_t* ea_bf  = (bf16_t*)alloc((size_t)E_EDGES * DIM * sizeof(bf16_t));
  bf16_t* Wct    = (bf16_t*)alloc((size_t)LAYERS * DIM * DIM * sizeof(bf16_t));
  float*  bc     = (float*) alloc((size_t)LAYERS * DIM * sizeof(float));
  bf16_t* Wlt    = (bf16_t*)alloc((size_t)LAYERS * DIM * DIM * sizeof(bf16_t));
  bf16_t* Wrt    = (bf16_t*)alloc((size_t)LAYERS * DIM * DIM * sizeof(bf16_t));
  bf16_t* Wlint  = (bf16_t*)alloc((size_t)LAYERS * DIM * DIM * sizeof(bf16_t));
  bf16_t* xbf    = (bf16_t*)alloc((size_t)N_NODES * DIM * sizeof(bf16_t));
  float*  xl     = (float*) alloc((size_t)N_NODES * DIM * sizeof(float));
  float*  xr     = (float*) alloc((size_t)N_NODES * DIM * sizeof(float));
  bf16_t* actbf  = (bf16_t*)alloc((size_t)N_NODES * DIM * sizeof(bf16_t));
  float*  hbuf   = (float*) alloc((size_t)N_NODES * DIM * sizeof(float));
  float*  logits = (float*) alloc((size_t)E_EDGES * 4 * sizeof(float));
  float*  exs    = (float*) alloc((size_t)E_EDGES * 4 * sizeof(float));
  // contiguous zero-zone: agg [N*128 f32] | denom [N*4 f32] | amax [N*4 u32]
  float* agg = (float*)alloc((size_t)N_NODES * (DIM + 8) * sizeof(float));
  float* denom = agg + (size_t)N_NODES * DIM;
  unsigned* amax = (unsigned*)(denom + (size_t)N_NODES * 4);

  const long nE = (long)E_EDGES * DIM;   // 81,920,000
  const long nX = (long)N_NODES * DIM;   // 2,560,000

  // prep
  k_f32_to_bf16<<<(int)((nE + 255) / 256), 256, 0, stream>>>(edge_attr, ea_bf, nE);
  k_f32_to_bf16<<<(int)((nX + 255) / 256), 256, 0, stream>>>(x, xbf, nX);
  k_prep_weights<<<(LAYERS * DIM * DIM + 255) / 256, 256, 0, stream>>>(
      Wee, bee, Wl, Wr, Wlin, We, Wct, bc, Wlt, Wrt, Wlint);

  for (int l = 0; l < LAYERS; ++l) {
    const long zoneQuads = (long)N_NODES * (DIM + 8) / 4;
    k_zero_u128<<<(int)((zoneQuads + 255) / 256), 256, 0, stream>>>((uint4*)agg, zoneQuads);
    k_gemm128<false><<<dim3(N_NODES / 16, 8), 32, 0, stream>>>(
        xbf, Wlt + (size_t)l * DIM * DIM, bl + l * DIM, xl, (bf16_t*)nullptr);
    k_gemm128<false><<<dim3(N_NODES / 16, 8), 32, 0, stream>>>(
        xbf, Wrt + (size_t)l * DIM * DIM, br + l * DIM, xr, (bf16_t*)nullptr);
    k_attn_logits<<<E_EDGES / 16, 32, 0, stream>>>(
        ea_bf, Wct + (size_t)l * DIM * DIM, bc + l * DIM, xl, xr, src, dst,
        att + l * DIM, logits, amax);
    k_attn_exp<<<(E_EDGES + 255) / 256, 256, 0, stream>>>(logits, dst, amax, denom, exs);
    k_attn_scatter<<<(E_EDGES * 4 + 255) / 256, 256, 0, stream>>>(exs, denom, xl, src, dst, agg);
    k_act<<<(int)((nX + 255) / 256), 256, 0, stream>>>(agg, cb + l * DIM, actbf);
    k_gemm128<true><<<dim3(N_NODES / 16, 8), 32, 0, stream>>>(
        actbf, Wlint + (size_t)l * DIM * DIM, blin + l * DIM, hbuf, xbf);
  }

  k_classifier<<<(N_NODES * NCLASS + 255) / 256, 256, 0, stream>>>(hbuf, Wout, bout, out);
}
